// PointnetSAModuleVotesWith_4209067950522
// MI455X (gfx1250) — matrix operand order repaired and checked by reading, not compiled, not run on hardware
//
#include <hip/hip_runtime.h>
#include <hip/hip_bf16.h>

// ---------------- problem constants ----------------
#define BATCH   4
#define NPTS    16384      // N
#define MFIX    256        // fixed_xyz count
#define CIN     128        // feature channels
#define NPOINT  2048
#define SQ      (MFIX + NPOINT)     // 2304 query points
#define NPALL   (MFIX + NPTS)       // 16640 all points
#define KS      64                  // nsample
#define ROWS    ((size_t)BATCH * SQ * KS)   // 589824 GEMM rows
#define LD0     160                 // 131 padded to mult of 32
#define O1      128
#define O2      128
#define O3      256
#define R2      0.04f
#define INV_R   5.0f
#define BN_EPS  1e-5f

typedef __attribute__((ext_vector_type(16))) _Float16 v16h;
typedef __attribute__((ext_vector_type(8)))  _Float16 v8h;
typedef __attribute__((ext_vector_type(8)))  float    v8f;

// ---------------- FPS: one block per batch ----------------
#define FPS_T 1024
#define FPS_P (NPTS / FPS_T)   // 16 points per thread

__global__ void fps_kernel(const float* __restrict__ xyz, int* __restrict__ inds) {
    const int b = blockIdx.x;
    const int t = threadIdx.x;
    __shared__ float sval[FPS_T];
    __shared__ int   sidx[FPS_T];
    __shared__ float lastx, lasty, lastz;

    float px[FPS_P], py[FPS_P], pz[FPS_P], dist[FPS_P];
    #pragma unroll
    for (int i = 0; i < FPS_P; ++i) {
        const int p = i * FPS_T + t;
        const size_t base = ((size_t)b * NPTS + p) * 3;
        px[i] = xyz[base + 0]; py[i] = xyz[base + 1]; pz[i] = xyz[base + 2];
        dist[i] = 1e10f;
    }
    if (t == 0) {
        const size_t base = (size_t)b * NPTS * 3;
        lastx = xyz[base]; lasty = xyz[base + 1]; lastz = xyz[base + 2];
        inds[b * NPOINT] = 0;
    }
    __syncthreads();

    for (int it = 1; it < NPOINT; ++it) {
        const float lx = lastx, ly = lasty, lz = lastz;
        float bv = -1.0f; int bi = 0x7fffffff;
        #pragma unroll
        for (int i = 0; i < FPS_P; ++i) {
            const float dx = px[i] - lx, dy = py[i] - ly, dz = pz[i] - lz;
            const float d = dx * dx + dy * dy + dz * dz;
            dist[i] = fminf(dist[i], d);
            const int gi = i * FPS_T + t;
            if (dist[i] > bv || (dist[i] == bv && gi < bi)) { bv = dist[i]; bi = gi; }
        }
        sval[t] = bv; sidx[t] = bi;
        __syncthreads();
        for (int s = FPS_T / 2; s > 0; s >>= 1) {
            if (t < s) {
                const float ov = sval[t + s]; const int oi = sidx[t + s];
                if (ov > sval[t] || (ov == sval[t] && oi < sidx[t])) { sval[t] = ov; sidx[t] = oi; }
            }
            __syncthreads();
        }
        if (t == 0) {
            const int w = sidx[0];
            inds[b * NPOINT + it] = w;
            const size_t base = ((size_t)b * NPTS + w) * 3;
            lastx = xyz[base]; lasty = xyz[base + 1]; lastz = xyz[base + 2];
        }
        __syncthreads();
    }
}

// ---------------- concat helpers ----------------
__global__ void allxyz_kernel(const float* __restrict__ fixed_xyz,
                              const float* __restrict__ xyz,
                              float* __restrict__ allxyz) {
    const int tid = blockIdx.x * blockDim.x + threadIdx.x;
    if (tid >= BATCH * NPALL) return;
    const int b = tid / NPALL, p = tid % NPALL;
    #pragma unroll
    for (int c = 0; c < 3; ++c) {
        const float v = (p < MFIX) ? fixed_xyz[((size_t)b * MFIX + p) * 3 + c]
                                   : xyz[((size_t)b * NPTS + (p - MFIX)) * 3 + c];
        allxyz[(size_t)tid * 3 + c] = v;
    }
}

__global__ void newxyz_kernel(const float* __restrict__ fixed_xyz,
                              const float* __restrict__ xyz,
                              const int* __restrict__ inds,
                              float* __restrict__ newxyz,
                              float* __restrict__ out_xyz,
                              float* __restrict__ out_inds) {
    const int tid = blockIdx.x * blockDim.x + threadIdx.x;
    if (tid < BATCH * NPOINT) out_inds[tid] = (float)inds[tid];
    if (tid >= BATCH * SQ) return;
    const int b = tid / SQ, s = tid % SQ;
    #pragma unroll
    for (int c = 0; c < 3; ++c) {
        float v;
        if (s < MFIX) v = fixed_xyz[((size_t)b * MFIX + s) * 3 + c];
        else          v = xyz[((size_t)b * NPTS + inds[b * NPOINT + (s - MFIX)]) * 3 + c];
        newxyz[(size_t)tid * 3 + c] = v;
        out_xyz[(size_t)tid * 3 + c] = v;
    }
}

// ---------------- ball query ----------------
__global__ void ballq_kernel(const float* __restrict__ allxyz,
                             const float* __restrict__ newxyz,
                             int* __restrict__ idx) {
    const int tid = blockIdx.x * blockDim.x + threadIdx.x;
    if (tid >= BATCH * SQ) return;
    const int b = tid / SQ;
    const float cx = newxyz[(size_t)tid * 3 + 0];
    const float cy = newxyz[(size_t)tid * 3 + 1];
    const float cz = newxyz[(size_t)tid * 3 + 2];
    const float* pts = allxyz + (size_t)b * NPALL * 3;
    int cnt = 0, first = -1;
    const size_t base = (size_t)tid * KS;
    for (int j = 0; j < NPALL; ++j) {
        const float dx = cx - pts[(size_t)j * 3 + 0];
        const float dy = cy - pts[(size_t)j * 3 + 1];
        const float dz = cz - pts[(size_t)j * 3 + 2];
        if (dx * dx + dy * dy + dz * dz < R2) {
            if (first < 0) first = j;
            idx[base + cnt] = j;
            if (++cnt == KS) break;
        }
    }
    const int pad = (first < 0) ? 0 : first;
    for (; cnt < KS; ++cnt) idx[base + cnt] = pad;
}

// ---------------- build X0 (f16, [ROWS][LD0]) ----------------
__global__ void buildx0_kernel(const float* __restrict__ allxyz,
                               const float* __restrict__ newxyz,
                               const float* __restrict__ feats,   // (B, C, NPALL)
                               const int* __restrict__ idx,
                               _Float16* __restrict__ X0) {
    const size_t tid = (size_t)blockIdx.x * blockDim.x + threadIdx.x;
    if (tid >= ROWS * LD0) return;
    const size_t m = tid / LD0;
    const int c = (int)(tid % LD0);
    const int sg = (int)(m / KS);            // b*SQ + s
    const int b = sg / SQ, s = sg % SQ;
    const int j = idx[m];
    float v;
    if (c < 3)
        v = (allxyz[((size_t)b * NPALL + j) * 3 + c] - newxyz[((size_t)b * SQ + s) * 3 + c]) * INV_R;
    else if (c < 131)
        v = feats[((size_t)b * CIN + (c - 3)) * NPALL + j];
    else
        v = 0.0f;
    X0[tid] = (_Float16)v;
}

// ---------------- weight convert (f32 [O][Kin] -> f16 [O][ldk], zero-pad) ----
__global__ void convw_kernel(const float* __restrict__ W, _Float16* __restrict__ Wh,
                             int O, int Kin, int ldk) {
    const int tid = blockIdx.x * blockDim.x + threadIdx.x;
    if (tid >= O * ldk) return;
    const int o = tid / ldk, c = tid % ldk;
    Wh[tid] = (_Float16)((c < Kin) ? W[(size_t)o * Kin + c] : 0.0f);
}

// ---------------- WMMA GEMM: Y[rows][O] = X[rows][ldk] * W[O][ldk]^T ---------
// Block = 256 threads (8 waves). The whole weight panel W[O][ldk] (f16) is
// staged into LDS once per block via GLOBAL_LOAD_ASYNC_TO_LDS_B128
// (ASYNCcnt path, ISA 08 §4), then each wave computes a 16(M)x64(N) strip.
__global__ void gemm_f16_wmma(const _Float16* __restrict__ X,
                              const _Float16* __restrict__ W,
                              float* __restrict__ Y,
                              int rows, int O, int ldk) {
    extern __shared__ _Float16 sW[];         // O * ldk halfs
    const int tid = threadIdx.x;

    // ---- async stage W -> LDS (16B per lane per op) ----
    {
        const int nbytes = O * ldk * 2;
        const unsigned ldsbase = (unsigned)(uintptr_t)&sW[0];   // AS3 offset = low 32 bits
        const char* gW = (const char*)W;
        for (int i = tid * 16; i < nbytes; i += 256 * 16) {
            const unsigned loff = ldsbase + (unsigned)i;
            const unsigned long long gaddr = (unsigned long long)(uintptr_t)(gW + i);
            asm volatile("global_load_async_to_lds_b128 %0, %1, off"
                         :: "v"(loff), "v"(gaddr) : "memory");
        }
        asm volatile("s_wait_asynccnt 0" ::: "memory");
        __syncthreads();
    }

    const int ngroups = O >> 6;              // 64 output cols per wave
    const int mtilesPerBlock = 8 / ngroups;
    const int wid = tid >> 5;
    const int lane = tid & 31;
    const int mtile = blockIdx.x * mtilesPerBlock + wid / ngroups;
    const int ng = wid % ngroups;
    if (mtile * 16 >= rows) return;

    const int half = lane >> 4;              // 0: lanes 0-15, 1: lanes 16-31
    const int l15 = lane & 15;
    const int m = mtile * 16 + l15;
    const int ncolbase = ng * 64;

    v8f acc[4] = {};
    const _Float16* __restrict__ xrow = X + (size_t)m * ldk;
    const int nchunks = ldk >> 5;

    for (int kc = 0; kc < nchunks; ++kc) {
        const int kb = kc * 32;
        __builtin_prefetch(xrow + kb + 32, 0, 1);   // global_prefetch_b8
        // A fragment: 16x32 f16 (ISA 7.12.2 layout) from global (streamed)
        v16h a;
        {
            const v8h lo = *(const v8h*)(xrow + kb + (half ? 8 : 0));
            const v8h hi = *(const v8h*)(xrow + kb + 16 + (half ? 8 : 0));
            #pragma unroll
            for (int i = 0; i < 8; ++i) { a[i] = lo[i]; a[8 + i] = hi[i]; }
        }
        #pragma unroll
        for (int t = 0; t < 4; ++t) {
            // B fragment: 32x16, B[k][n] = W[n][k]; lane n holds k 8h..8h+7 and 16+8h..16+8h+7 (from LDS)
            const int n = ncolbase + t * 16 + l15;
            const _Float16* wrow = sW + (size_t)n * ldk + kb + (half ? 8 : 0);
            v16h bf;
            const v8h b0 = *(const v8h*)(wrow);
            const v8h b1 = *(const v8h*)(wrow + 16);
            #pragma unroll
            for (int i = 0; i < 8; ++i) { bf[i] = b0[i]; bf[8 + i] = b1[i]; }
            acc[t] = __builtin_amdgcn_wmma_f32_16x16x32_f16(
                false, a, false, bf, (short)0, acc[t], false, false);
        }
    }
    // D store: VGPR i -> row mtile*16 + i + 8*half, col = ncolbase + t*16 + l15
    #pragma unroll
    for (int t = 0; t < 4; ++t) {
        const int n = ncolbase + t * 16 + l15;
        #pragma unroll
        for (int i = 0; i < 8; ++i) {
            const int mr = mtile * 16 + i + (half ? 8 : 0);
            Y[(size_t)mr * O + n] = acc[t][i];
        }
    }
}

// ---------------- BN stats / apply / fused max ----------------
__global__ void zero_kernel(float* p, int n) {
    const int tid = blockIdx.x * blockDim.x + threadIdx.x;
    if (tid < n) p[tid] = 0.0f;
}

#define SLICES 4096
__global__ void bn_stats_kernel(const float* __restrict__ Y, float* __restrict__ stats,
                                int rows, int O) {
    const int tid = blockIdx.x * blockDim.x + threadIdx.x;
    if (tid >= O * SLICES) return;
    const int o = tid % O;
    const int slice = tid / O;
    float s = 0.0f, ss = 0.0f;
    for (int r = slice; r < rows; r += SLICES) {
        const float v = Y[(size_t)r * O + o];
        s += v; ss += v * v;
    }
    atomicAdd(&stats[o], s);
    atomicAdd(&stats[256 + o], ss);
}

__global__ void bn_apply_kernel(const float* __restrict__ Y, const float* __restrict__ stats,
                                const float* __restrict__ g, const float* __restrict__ bb,
                                _Float16* __restrict__ Xn, int rows, int O) {
    const size_t tid = (size_t)blockIdx.x * blockDim.x + threadIdx.x;
    if (tid >= (size_t)rows * O) return;
    const int o = (int)(tid % O);
    const float inv = 1.0f / (float)rows;
    const float mean = stats[o] * inv;
    const float var = stats[256 + o] * inv - mean * mean;
    float v = (Y[tid] - mean) * rsqrtf(var + BN_EPS) * g[o] + bb[o];
    Xn[tid] = (_Float16)fmaxf(v, 0.0f);
}

__global__ void bn_max_kernel(const float* __restrict__ Y, const float* __restrict__ stats,
                              const float* __restrict__ g, const float* __restrict__ bb,
                              float* __restrict__ out_feat, int rows) {
    const int tid = blockIdx.x * blockDim.x + threadIdx.x;
    if (tid >= BATCH * SQ * O3) return;
    const int o = tid % O3;
    const int sg = tid / O3;           // b*SQ + s
    const int b = sg / SQ, s = sg % SQ;
    const float inv = 1.0f / (float)rows;
    const float mean = stats[o] * inv;
    const float var = stats[256 + o] * inv - mean * mean;
    const float scale = rsqrtf(var + BN_EPS) * g[o];
    const float bias = bb[o];
    float mx = 0.0f;                   // relu outputs are >= 0
    for (int k = 0; k < KS; ++k) {
        float v = (Y[((size_t)sg * KS + k) * O3 + o] - mean) * scale + bias;
        mx = fmaxf(mx, fmaxf(v, 0.0f));
    }
    out_feat[((size_t)b * O3 + o) * SQ + s] = mx;
}

// ---------------- host ----------------
static inline size_t align256(size_t x) { return (x + 255) & ~(size_t)255; }

extern "C" void kernel_launch(void* const* d_in, const int* in_sizes, int n_in,
                              void* d_out, int out_size, void* d_ws, size_t ws_size,
                              hipStream_t stream) {
    const float* fixed_xyz = (const float*)d_in[0];
    const float* xyz       = (const float*)d_in[1];
    const float* feats     = (const float*)d_in[2];
    const float* W1 = (const float*)d_in[3];
    const float* g1 = (const float*)d_in[4];
    const float* b1 = (const float*)d_in[5];
    const float* W2 = (const float*)d_in[6];
    const float* g2 = (const float*)d_in[7];
    const float* b2 = (const float*)d_in[8];
    const float* W3 = (const float*)d_in[9];
    const float* g3 = (const float*)d_in[10];
    const float* b3 = (const float*)d_in[11];

    // workspace layout
    char* ws = (char*)d_ws;
    size_t off = 0;
    int* inds = (int*)(ws + off);            off = align256(off + (size_t)BATCH * NPOINT * 4);
    float* allxyz = (float*)(ws + off);      off = align256(off + (size_t)BATCH * NPALL * 3 * 4);
    float* newxyz = (float*)(ws + off);      off = align256(off + (size_t)BATCH * SQ * 3 * 4);
    int* idx = (int*)(ws + off);             off = align256(off + ROWS * 4);
    float* stats = (float*)(ws + off);       off = align256(off + 512 * 4);
    _Float16* Wh1 = (_Float16*)(ws + off);   off = align256(off + (size_t)O1 * LD0 * 2);
    _Float16* Wh2 = (_Float16*)(ws + off);   off = align256(off + (size_t)O2 * O1 * 2);
    _Float16* Wh3 = (_Float16*)(ws + off);   off = align256(off + (size_t)O3 * O2 * 2);
    _Float16* Xb = (_Float16*)(ws + off);    off = align256(off + ROWS * LD0 * 2);   // X0/X1/X2
    float* Yb = (float*)(ws + off);          off = align256(off + ROWS * O3 * 4);    // Y1/Y2/Y3

    // d_out layout: new_xyz | new_features | inds(as float)
    float* out_xyz  = (float*)d_out;
    float* out_feat = out_xyz + (size_t)BATCH * SQ * 3;
    float* out_inds = out_feat + (size_t)BATCH * O3 * SQ;

    // 1. FPS
    fps_kernel<<<BATCH, FPS_T, 0, stream>>>(xyz, inds);
    // 2. concat point sets
    allxyz_kernel<<<(BATCH * NPALL + 255) / 256, 256, 0, stream>>>(fixed_xyz, xyz, allxyz);
    newxyz_kernel<<<(BATCH * SQ + 255) / 256, 256, 0, stream>>>(fixed_xyz, xyz, inds,
                                                                newxyz, out_xyz, out_inds);
    // 3. ball query
    ballq_kernel<<<(BATCH * SQ + 127) / 128, 128, 0, stream>>>(allxyz, newxyz, idx);
    // 4. gather + concat -> X0 (f16)
    {
        const size_t total = ROWS * LD0;
        buildx0_kernel<<<(unsigned)((total + 255) / 256), 256, 0, stream>>>(
            allxyz, newxyz, feats, idx, Xb);
    }
    // 5. weights -> f16 (padded)
    convw_kernel<<<(O1 * LD0 + 255) / 256, 256, 0, stream>>>(W1, Wh1, O1, 131, LD0);
    convw_kernel<<<(O2 * O1 + 255) / 256, 256, 0, stream>>>(W2, Wh2, O2, O1, O1);
    convw_kernel<<<(O3 * O2 + 255) / 256, 256, 0, stream>>>(W3, Wh3, O3, O2, O2);

    const int mtiles = (int)(ROWS / 16);
    // ---- layer 1: [ROWS][160] x [128][160]^T
    {
        const unsigned blocks = (unsigned)(mtiles / (8 / (O1 >> 6)));
        gemm_f16_wmma<<<blocks, 256, O1 * LD0 * 2, stream>>>(Xb, Wh1, Yb, (int)ROWS, O1, LD0);
        zero_kernel<<<2, 256, 0, stream>>>(stats, 512);
        bn_stats_kernel<<<(O1 * SLICES + 255) / 256, 256, 0, stream>>>(Yb, stats, (int)ROWS, O1);
        bn_apply_kernel<<<(unsigned)((ROWS * O1 + 255) / 256), 256, 0, stream>>>(
            Yb, stats, g1, b1, Xb, (int)ROWS, O1);
    }
    // ---- layer 2: [ROWS][128] x [128][128]^T
    {
        const unsigned blocks = (unsigned)(mtiles / (8 / (O2 >> 6)));
        gemm_f16_wmma<<<blocks, 256, O2 * O1 * 2, stream>>>(Xb, Wh2, Yb, (int)ROWS, O2, O1);
        zero_kernel<<<2, 256, 0, stream>>>(stats, 512);
        bn_stats_kernel<<<(O2 * SLICES + 255) / 256, 256, 0, stream>>>(Yb, stats, (int)ROWS, O2);
        bn_apply_kernel<<<(unsigned)((ROWS * O2 + 255) / 256), 256, 0, stream>>>(
            Yb, stats, g2, b2, Xb, (int)ROWS, O2);
    }
    // ---- layer 3: [ROWS][128] x [256][128]^T, fused BN+ReLU+max -> out
    {
        const unsigned blocks = (unsigned)(mtiles / (8 / (O3 >> 6)));
        gemm_f16_wmma<<<blocks, 256, O3 * O2 * 2, stream>>>(Xb, Wh3, Yb, (int)ROWS, O3, O2);
        zero_kernel<<<2, 256, 0, stream>>>(stats, 512);
        bn_stats_kernel<<<(O3 * SLICES + 255) / 256, 256, 0, stream>>>(Yb, stats, (int)ROWS, O3);
        bn_max_kernel<<<(BATCH * SQ * O3 + 255) / 256, 256, 0, stream>>>(
            Yb, stats, g3, b3, out_feat, (int)ROWS);
    }
    (void)in_sizes; (void)n_in; (void)out_size; (void)ws_size;
}
